// TrajectoryAutoencoder_16458314678719
// MI455X (gfx1250) — compile-verified, matrix operand order checked by reading, not run on hardware
//
#include <hip/hip_runtime.h>
#include <math.h>

typedef __attribute__((ext_vector_type(16))) _Float16 v16h;
typedef __attribute__((ext_vector_type(8)))  float    v8f;

#define Bv   64
#define Tv   256
#define Dv   64
#define Hv   512
#define G4H  2048
#define MAX_KTILES 34   // K up to 1088 -> 34 k-tiles of 32

__device__ __forceinline__ float sigf(float x) { return 1.0f / (1.0f + __expf(-x)); }

// Map element e (0..15) + lane (0..31) -> k_local within a 32-wide K tile.
// ISA 16-bit A-matrix 16x32 layout; mirrored map used for B (lane%16 = N).
__device__ __forceinline__ int klocal_of(int lane, int e) {
    int half = lane >> 4;
    int vg = e >> 1, p = e & 1;
    return (vg < 4 ? 0 : 16) + half * 8 + ((vg & 3) << 1) + p;
}

// -------- Pack fp32 weight rows (z = a @ W^T) into f16 WMMA-B tile layout ----
// out layout: [tk][tn][lane(32)][e(16)] f16, tile = 1KB.
__global__ void pack_weight(const float* __restrict__ p0, int s0, int w0,
                            const float* __restrict__ p1, int s1, int w1,
                            _Float16* __restrict__ out, int N, int K)
{
    int nTileN = N >> 4;
    int total = (K >> 5) * nTileN * 512;
    for (int idx = blockIdx.x * blockDim.x + threadIdx.x; idx < total;
         idx += gridDim.x * blockDim.x) {
        int e    = idx & 15;
        int lane = (idx >> 4) & 31;
        int tn   = (idx >> 9) % nTileN;
        int tk   = idx / (nTileN << 9);
        int k = tk * 32 + klocal_of(lane, e);
        int n = tn * 16 + (lane & 15);
        float v = 0.f;
        if (k < w0)               v = p0[(size_t)n * s0 + k];
        else if ((k - w0) < w1)   v = p1[(size_t)n * s1 + (k - w0)];
        out[idx] = (_Float16)v;
    }
}

// -------- WMMA GEMM: C[M,N] = concat(A0,A1,A2)[M,K] * Wp^T + bias ------------
// One workgroup owns a 64-row M-slab (4 m-tiles) and 8 n-tiles (one per wave).
// Full A slab staged in LDS (f16, WMMA-A layout, up to 136KB). Each wave holds
// 4 accumulators; inner loop is software-pipelined with the last iteration
// peeled, so next k-tile's loads issue unconditionally (no selects) and
// overlap the current 4-WMMA burst.
__global__ void wmma_gemm(const float* __restrict__ a0, int s0, int w0,
                          const float* __restrict__ a1, int s1, int w1,
                          const float* __restrict__ a2, int s2, int w2,
                          const _Float16* __restrict__ Wp,
                          const float* __restrict__ bias,
                          float* __restrict__ C, int ldC,
                          int M, int N, int K)
{
    __shared__ _Float16 alds[4 * MAX_KTILES * 512];   // [mi][tk][lane][e]
    const int nTileK  = K >> 5;
    const int nTileN  = N >> 4;
    const int nWGperM = (nTileN + 7) >> 3;
    const int mt4 = blockIdx.x / nWGperM;             // 64-row slab index
    const int wgN = blockIdx.x % nWGperM;
    const int tid = threadIdx.x;
    const int w01 = w0 + w1, w012 = w01 + w2;

    // stage A slab (64 rows x K) into LDS, fp32 -> f16, WMMA-A layout
    const int slab = nTileK * 512;
    for (int i = tid; i < 4 * slab; i += blockDim.x) {
        int e    = i & 15;
        int lane = (i >> 4) & 31;
        int tk   = (i >> 9) % nTileK;
        int mi   = i / slab;
        int k = tk * 32 + klocal_of(lane, e);
        int m = mt4 * 64 + mi * 16 + (lane & 15);
        float v = 0.f;
        if (k < w0)        v = a0[(size_t)m * s0 + k];
        else if (k < w01)  v = a1[(size_t)m * s1 + (k - w0)];
        else if (k < w012) v = a2[(size_t)m * s2 + (k - w01)];
        alds[i] = (_Float16)v;
    }
    __syncthreads();

    const int wave = tid >> 5;
    const int lane = tid & 31;
    const int tn = wgN * 8 + wave;
    if (tn >= nTileN) return;                 // wave-uniform: EXEC stays full

    v8f acc0 = {}, acc1 = {}, acc2 = {}, acc3 = {};
    const size_t wstride = (size_t)nTileN * 512;
    const _Float16* wp = Wp + ((size_t)tn * 32 + lane) * 16;
    // per-m-tile LDS base pointers (constant-stride ds addressing in the loop)
    const _Float16* ap0 = &alds[0 * slab + lane * 16];
    const _Float16* ap1 = &alds[1 * slab + lane * 16];
    const _Float16* ap2 = &alds[2 * slab + lane * 16];
    const _Float16* ap3 = &alds[3 * slab + lane * 16];

    // prologue: load k-tile 0
    v16h bcur = *reinterpret_cast<const v16h*>(wp);
    v16h A0 = *reinterpret_cast<const v16h*>(ap0);
    v16h A1 = *reinterpret_cast<const v16h*>(ap1);
    v16h A2 = *reinterpret_cast<const v16h*>(ap2);
    v16h A3 = *reinterpret_cast<const v16h*>(ap3);

    for (int tk = 0; tk + 1 < nTileK; ++tk) {
        // fetch k-tile tk+1 into a second register set (unconditional)
        wp  += wstride;
        ap0 += 512; ap1 += 512; ap2 += 512; ap3 += 512;
        v16h bnext = *reinterpret_cast<const v16h*>(wp);
        __builtin_prefetch(wp + wstride, 0, 3);       // global_prefetch_b8
        v16h N0 = *reinterpret_cast<const v16h*>(ap0);
        v16h N1 = *reinterpret_cast<const v16h*>(ap1);
        v16h N2 = *reinterpret_cast<const v16h*>(ap2);
        v16h N3 = *reinterpret_cast<const v16h*>(ap3);

        acc0 = __builtin_amdgcn_wmma_f32_16x16x32_f16(false, A0, false, bcur, (short)0, acc0, false, false);
        acc1 = __builtin_amdgcn_wmma_f32_16x16x32_f16(false, A1, false, bcur, (short)0, acc1, false, false);
        acc2 = __builtin_amdgcn_wmma_f32_16x16x32_f16(false, A2, false, bcur, (short)0, acc2, false, false);
        acc3 = __builtin_amdgcn_wmma_f32_16x16x32_f16(false, A3, false, bcur, (short)0, acc3, false, false);

        bcur = bnext; A0 = N0; A1 = N1; A2 = N2; A3 = N3;
    }
    // epilogue: final k-tile
    acc0 = __builtin_amdgcn_wmma_f32_16x16x32_f16(false, A0, false, bcur, (short)0, acc0, false, false);
    acc1 = __builtin_amdgcn_wmma_f32_16x16x32_f16(false, A1, false, bcur, (short)0, acc1, false, false);
    acc2 = __builtin_amdgcn_wmma_f32_16x16x32_f16(false, A2, false, bcur, (short)0, acc2, false, false);
    acc3 = __builtin_amdgcn_wmma_f32_16x16x32_f16(false, A3, false, bcur, (short)0, acc3, false, false);

    const int mloc = (lane >> 4) << 3;          // 0 or 8
    const int n = tn * 16 + (lane & 15);
    const float bb = bias ? bias[n] : 0.f;
    v8f accs[4] = {acc0, acc1, acc2, acc3};
#pragma unroll
    for (int mi = 0; mi < 4; ++mi) {
#pragma unroll
        for (int r = 0; r < 8; ++r) {
            int m = mt4 * 64 + mi * 16 + mloc + r;
            C[(size_t)m * ldC + n] = accs[mi][r] + bb;
        }
    }
}

// -------- LSTM gate nonlinearity: consumes z (B x 4H), updates h,c -----------
__global__ void lstm_gates(const float* __restrict__ z,
                           float* __restrict__ h, float* __restrict__ c,
                           float* __restrict__ seq, int ldSeq)
{
    int idx = blockIdx.x * blockDim.x + threadIdx.x;
    if (idx >= Bv * Hv) return;
    int b = idx / Hv, j = idx % Hv;
    const float* zr = z + (size_t)b * G4H;
    float zi = zr[j], zf = zr[Hv + j], zg = zr[2 * Hv + j], zo = zr[3 * Hv + j];
    float cn = sigf(zf) * c[idx] + sigf(zi) * tanhf(zg);
    c[idx] = cn;
    float hn = sigf(zo) * tanhf(cn);
    h[idx] = hn;
    if (seq) seq[(size_t)b * ldSeq + j] = hn;
}

// -------- attention scores: s[b,t] = sum_h v[h]*tanh(enc_proj[b,t,h]+hW[b,h])-
__global__ void attn_scores(const float* __restrict__ enc_proj,
                            const float* __restrict__ hW,
                            const float* __restrict__ vvec,
                            float* __restrict__ scores)
{
    int row  = blockIdx.x * 8 + (threadIdx.x >> 5);   // (b*T + t)
    int lane = threadIdx.x & 31;
    int b = row / Tv;
    const float* ep = enc_proj + (size_t)row * Hv;
    const float* hw = hW + (size_t)b * Hv;
    float s = 0.f;
    for (int h = lane; h < Hv; h += 32)
        s += vvec[h] * tanhf(ep[h] + hw[h]);
    for (int off = 16; off > 0; off >>= 1)
        s += __shfl_xor(s, off, 32);
    if (lane == 0) scores[row] = s;
}

// -------- softmax over T + context = attn-weighted sum of enc_out -----------
__global__ void attn_softmax_context(const float* __restrict__ scores,
                                     const float* __restrict__ enc_out,
                                     float* __restrict__ context)
{
    __shared__ float red[256];
    __shared__ float att[256];
    int b = blockIdx.x, t = threadIdx.x;
    float s = scores[b * Tv + t];
    red[t] = s; __syncthreads();
    for (int st = 128; st > 0; st >>= 1) {
        if (t < st) red[t] = fmaxf(red[t], red[t + st]);
        __syncthreads();
    }
    float mx = red[0]; __syncthreads();
    float e = __expf(s - mx);
    red[t] = e; __syncthreads();
    for (int st = 128; st > 0; st >>= 1) {
        if (t < st) red[t] += red[t + st];
        __syncthreads();
    }
    att[t] = e / red[0];
    __syncthreads();
    const float* eo = enc_out + (size_t)b * Tv * Hv;
    for (int h = t; h < Hv; h += 256) {
        float acc = 0.f;
        for (int tt = 0; tt < Tv; ++tt)
            acc += att[tt] * eo[(size_t)tt * Hv + h];
        context[b * Hv + h] = acc;
    }
}

__global__ void zero_f32(float* p, int n)
{
    int i = blockIdx.x * blockDim.x + threadIdx.x;
    if (i < n) p[i] = 0.f;
}

extern "C" void kernel_launch(void* const* d_in, const int* in_sizes, int n_in,
                              void* d_out, int out_size, void* d_ws, size_t ws_size,
                              hipStream_t stream)
{
    (void)in_sizes; (void)n_in; (void)out_size; (void)ws_size;
    const float* x      = (const float*)d_in[0];
    const float* eW_ih0 = (const float*)d_in[1];
    const float* eW_hh0 = (const float*)d_in[2];
    const float* eb0    = (const float*)d_in[3];
    const float* eW_ih1 = (const float*)d_in[4];
    const float* eW_hh1 = (const float*)d_in[5];
    const float* eb1    = (const float*)d_in[6];
    const float* dW_ih0 = (const float*)d_in[7];
    const float* dW_hh0 = (const float*)d_in[8];
    const float* db0    = (const float*)d_in[9];
    const float* dW_ih1 = (const float*)d_in[10];
    const float* dW_hh1 = (const float*)d_in[11];
    const float* db1    = (const float*)d_in[12];
    const float* attn_W = (const float*)d_in[13];
    const float* attn_b = (const float*)d_in[14];
    const float* vvec   = (const float*)d_in[15];
    const float* fc_W   = (const float*)d_in[16];
    const float* fc_b   = (const float*)d_in[17];
    float* out = (float*)d_out;

    char* ws = (char*)d_ws;
    size_t off = 0;
    auto carve = [&](size_t bytes) -> void* {
        size_t o = (off + 255) & ~(size_t)255;
        off = o + bytes;
        return (void*)(ws + o);
    };

    // packed f16 weights
    _Float16* pW0e  = (_Float16*)carve((size_t)576  * 2048 * 2);
    _Float16* pW1e  = (_Float16*)carve((size_t)1024 * 2048 * 2);
    _Float16* pWpj  = (_Float16*)carve((size_t)512  * 512  * 2);
    _Float16* pWh   = (_Float16*)carve((size_t)512  * 512  * 2);
    _Float16* pW0d  = (_Float16*)carve((size_t)1088 * 2048 * 2);
    _Float16* pW1d  = (_Float16*)carve((size_t)1024 * 2048 * 2);
    _Float16* pWfc  = (_Float16*)carve((size_t)512  * 64   * 2);
    // fp32 scratch
    float* z_buf    = (float*)carve((size_t)Bv * G4H * 4);
    float* states   = (float*)carve(((size_t)4 * Bv * Hv + Bv * Dv) * 4);
    float* h0 = states;
    float* c0 = h0 + Bv * Hv;
    float* h1 = c0 + Bv * Hv;
    float* c1 = h1 + Bv * Hv;
    float* zero_in = c1 + Bv * Hv;             // B x D zeros (initial dec_in)
    float* enc_out  = (float*)carve((size_t)Bv * Tv * Hv * 4);
    float* enc_proj = (float*)carve((size_t)Bv * Tv * Hv * 4);
    float* hW       = (float*)carve((size_t)Bv * Hv * 4);
    float* scores   = (float*)carve((size_t)Bv * Tv * 4);
    float* context  = (float*)carve((size_t)Bv * Hv * 4);

    auto pack = [&](const float* p0, int s0, int w0,
                    const float* p1, int s1, int w1,
                    _Float16* dst, int N, int K) {
        pack_weight<<<2048, 256, 0, stream>>>(p0, s0, w0, p1, s1, w1, dst, N, K);
    };
    auto gemm = [&](const float* A0, int S0, int W0,
                    const float* A1, int S1, int W1,
                    const float* A2, int S2, int W2,
                    const _Float16* Wp, const float* bias,
                    float* C, int ldC, int M, int N, int K) {
        int nTileN = N >> 4;
        int nWGperM = (nTileN + 7) >> 3;
        wmma_gemm<<<dim3((M >> 6) * nWGperM), 256, 0, stream>>>(
            A0, S0, W0, A1, S1, W1, A2, S2, W2, Wp, bias, C, ldC, M, N, K);
    };

    // ---- weight packing (every call; deterministic) ----
    pack(eW_ih0, Dv, Dv,       eW_hh0, Hv, Hv, pW0e, G4H, 576);
    pack(eW_ih1, Hv, Hv,       eW_hh1, Hv, Hv, pW1e, G4H, 1024);
    pack(attn_W + Hv, 2 * Hv, Hv, nullptr, 0, 0, pWpj, Hv, Hv);   // W_e
    pack(attn_W,      2 * Hv, Hv, nullptr, 0, 0, pWh,  Hv, Hv);   // W_h
    pack(dW_ih0, Dv + Hv, Dv + Hv, dW_hh0, Hv, Hv, pW0d, G4H, 1088);
    pack(dW_ih1, Hv, Hv,       dW_hh1, Hv, Hv, pW1d, G4H, 1024);
    pack(fc_W,   Hv, Hv,       nullptr, 0, 0,  pWfc, Dv, Hv);

    // ---- zero initial states + zero dec_in ----
    {
        int n = 4 * Bv * Hv + Bv * Dv;
        zero_f32<<<(n + 255) / 256, 256, 0, stream>>>(states, n);
    }

    // ---- encoder: 256 sequential steps ----
    for (int t = 0; t < Tv; ++t) {
        gemm(x + (size_t)t * Dv, Tv * Dv, Dv,   h0, Hv, Hv,   nullptr, 0, 0,
             pW0e, eb0, z_buf, G4H, Bv, G4H, 576);
        lstm_gates<<<(Bv * Hv + 255) / 256, 256, 0, stream>>>(z_buf, h0, c0, nullptr, 0);
        gemm(h0, Hv, Hv,   h1, Hv, Hv,   nullptr, 0, 0,
             pW1e, eb1, z_buf, G4H, Bv, G4H, 1024);
        lstm_gates<<<(Bv * Hv + 255) / 256, 256, 0, stream>>>(
            z_buf, h1, c1, enc_out + (size_t)t * Hv, Tv * Hv);
    }

    // ---- enc_proj = enc_out @ W_e^T + attn_b : one big parallel WMMA GEMM ----
    gemm(enc_out, Hv, Hv, nullptr, 0, 0, nullptr, 0, 0,
         pWpj, attn_b, enc_proj, Hv, Bv * Tv, Hv, Hv);

    // ---- decoder: 256 sequential steps with attention ----
    for (int t = 0; t < Tv; ++t) {
        gemm(h1, Hv, Hv, nullptr, 0, 0, nullptr, 0, 0,
             pWh, nullptr, hW, Hv, Bv, Hv, Hv);
        attn_scores<<<(Bv * Tv) / 8, 256, 0, stream>>>(enc_proj, hW, vvec, scores);
        attn_softmax_context<<<Bv, 256, 0, stream>>>(scores, enc_out, context);

        const float* dec_in = (t == 0) ? zero_in : (out + (size_t)(t - 1) * Dv);
        int dec_stride = (t == 0) ? Dv : Tv * Dv;
        gemm(dec_in, dec_stride, Dv,   context, Hv, Hv,   h0, Hv, Hv,
             pW0d, db0, z_buf, G4H, Bv, G4H, 1088);
        lstm_gates<<<(Bv * Hv + 255) / 256, 256, 0, stream>>>(z_buf, h0, c0, nullptr, 0);
        gemm(h0, Hv, Hv,   h1, Hv, Hv,   nullptr, 0, 0,
             pW1d, db1, z_buf, G4H, Bv, G4H, 1024);
        lstm_gates<<<(Bv * Hv + 255) / 256, 256, 0, stream>>>(z_buf, h1, c1, nullptr, 0);

        // out[:, t, :] = h1 @ fc_W^T + fc_b   (also feeds next step's dec_in)
        gemm(h1, Hv, Hv, nullptr, 0, 0, nullptr, 0, 0,
             pWfc, fc_b, out + (size_t)t * Dv, Tv * Dv, Bv, Dv, Hv);
    }
}
